// Qwen3OmniMoeTalkerSparseMoeBlock_76081050681724
// MI455X (gfx1250) — compile-verified, matrix operand order checked by reading, not run on hardware
//
#include <hip/hip_runtime.h>
#include <cstdint>
#include <cstddef>

#define H     1024
#define E_EXP 32
#define TOPK  4
#define I_RT  768
#define I_SH  2048
#define T_TOK 1024

typedef __attribute__((ext_vector_type(16))) __bf16 v16bf;
typedef __attribute__((ext_vector_type(4)))  __bf16 v4bf;
typedef __attribute__((ext_vector_type(8)))  float  v8f;

union Frag { v16bf v; uint4 q[2]; };

// -------------------- router: softmax + top-4 + sigmoid shared gate --------------------
__global__ __launch_bounds__(256) void router_kernel(
    const float* __restrict__ x, const float* __restrict__ rw,
    const float* __restrict__ shgw,
    float* __restrict__ topk_w, int* __restrict__ topk_idx,
    float* __restrict__ sig_gate)
{
    const int wave = threadIdx.x >> 5;
    const int lane = threadIdx.x & 31;
    const int t = blockIdx.x * 8 + wave;
    const float* xr = x + (size_t)t * H;

    float acc = 0.f, sacc = 0.f;
    for (int h = 0; h < H; h += 4) {
        const float4 xv = *(const float4*)(xr + h);
        acc += xv.x * rw[(h + 0) * E_EXP + lane];
        acc += xv.y * rw[(h + 1) * E_EXP + lane];
        acc += xv.z * rw[(h + 2) * E_EXP + lane];
        acc += xv.w * rw[(h + 3) * E_EXP + lane];
    }
    for (int h = lane; h < H; h += 32) sacc += xr[h] * shgw[h];
    for (int off = 16; off >= 1; off >>= 1) sacc += __shfl_xor(sacc, off, 32);

    // softmax over the 32 experts (one per lane)
    float m = acc;
    for (int off = 16; off >= 1; off >>= 1) m = fmaxf(m, __shfl_xor(m, off, 32));
    float p = __expf(acc - m);
    float s = p;
    for (int off = 16; off >= 1; off >>= 1) s += __shfl_xor(s, off, 32);
    p /= s;

    // iterative top-4 argmax (lowest-lane tie-break), then renormalize
    float v = p;
    float wsum = 0.f, myw = 0.f;
    int myi = 0;
    #pragma unroll
    for (int k = 0; k < TOPK; ++k) {
        float bv = v; int bl = lane;
        for (int off = 16; off >= 1; off >>= 1) {
            float ov = __shfl_xor(bv, off, 32);
            int   ol = __shfl_xor(bl, off, 32);
            if (ov > bv || (ov == bv && ol < bl)) { bv = ov; bl = ol; }
        }
        if (lane == bl) v = -1.f;
        wsum += bv;
        if (lane == k) { myw = bv; myi = bl; }
    }
    if (lane < TOPK) {
        topk_w[t * TOPK + lane] = myw / wsum;
        topk_idx[t * TOPK + lane] = myi;
    }
    if (lane == 0) sig_gate[t] = 1.f / (1.f + __expf(-sacc));
}

// -------------------- dispatch: build per-expert token/slot lists --------------------
__global__ __launch_bounds__(256) void dispatch_kernel(
    const int* __restrict__ topk_idx,
    int* __restrict__ cnt,
    int* __restrict__ list_tok,
    int* __restrict__ list_slot)
{
    const int i = blockIdx.x * blockDim.x + threadIdx.x;  // 0 .. T*4-1
    if (i >= T_TOK * TOPK) return;
    const int e = topk_idx[i];
    const int j = atomicAdd(&cnt[e], 1);
    list_tok[e * T_TOK + j]  = i >> 2;   // token id
    list_slot[e * T_TOK + j] = i;        // output slot = t*4+k
}

// -------------------- unified bf16-WMMA GEMM, templated on dual-B SwiGLU --------------------
#define BM 64
#define BN 128
#define BK 32
#define KP 40   // padded K stride in LDS (elements)

// XOR-swizzle of the 8-element K-groups within an LDS row (spreads bank access
// for the transposed b64 stores; applied identically at store and fragment load)
__device__ __forceinline__ int kswz(int row, int kByteGroup /*0,8,16,24*/) {
    const int g  = kByteGroup >> 3;       // 0..3
    const int gs = (row >> 4) & 3;
    return (g ^ gs) << 3;
}

template <int DUAL>
__global__ __launch_bounds__(256) void gemm_kernel(
    const float* __restrict__ A, int lda,
    const float* __restrict__ B1, const float* __restrict__ B2, int ldb,
    float* __restrict__ C, int ldc,
    size_t bExpStride,
    const int* __restrict__ rowInList, const int* __restrict__ rowOutList, int listStride,
    const int* __restrict__ Mcnt, int Mfixed,
    int K,
    const float* __restrict__ rowScale)
{
    __shared__ alignas(16) __bf16 sA [2 * BM * KP];
    __shared__ alignas(16) __bf16 sB1[2 * BN * KP];
    __shared__ alignas(16) __bf16 sB2[DUAL ? 2 * BN * KP : 1];
    __shared__ int s_in[BM];
    __shared__ int s_out[BM];

    const int e = blockIdx.z;
    const float* Bp1 = B1 + (size_t)e * bExpStride;
    const float* Bp2 = DUAL ? (B2 + (size_t)e * bExpStride) : nullptr;
    const int M = Mcnt ? Mcnt[e] : Mfixed;
    const int m0 = blockIdx.y * BM;
    if (m0 >= M) return;                        // empty tile for this expert
    const int n0 = blockIdx.x * BN;

    const int tid = threadIdx.x;
    if (tid < BM) {
        const int r = m0 + tid;
        int inr = 0, outr = -1;
        if (r < M) {
            inr  = rowInList  ? rowInList[e * listStride + r]  : r;
            outr = rowOutList ? rowOutList[e * listStride + r] : r;
        }
        s_in[tid] = inr;
        s_out[tid] = outr;
    }
    __syncthreads();

    const int lane = tid & 31;
    const int wv   = tid >> 5;
    const int wm0  = (wv >> 2) * 32;   // wave M origin: 0 / 32
    const int wn0  = (wv & 3) * 32;    // wave N origin: 0/32/64/96

    v8f accG[2][2], accU[2][2];
    #pragma unroll
    for (int i = 0; i < 2; ++i)
        #pragma unroll
        for (int j = 0; j < 2; ++j)
            #pragma unroll
            for (int r = 0; r < 8; ++r) { accG[i][j][r] = 0.f; accU[i][j][r] = 0.f; }

    const int arow = tid >> 3;            // 0..31
    const int acol = (tid & 7) * 4;       // 0..28
    const int bkq  = (tid >> 5) * 4;      // 0..28 (4 consecutive k rows)
    const int bn4  = (tid & 31) * 4;      // 0..124 (4 consecutive n cols)
    const int kb   = (lane < 16) ? 0 : 8; // ISA 16-bit A/B lane K-group
    const int lm   = lane & 15;

    // register staging buffers for the next K-chunk (software pipeline)
    float4 ra[2], rb1[4], rb2[DUAL ? 4 : 1];

    auto load_regs = [&](int kc) {
        #pragma unroll
        for (int it = 0; it < 2; ++it) {
            const int row = it * 32 + arow;
            ra[it] = *(const float4*)(A + (size_t)s_in[row] * lda + kc + acol);
        }
        #pragma unroll
        for (int r = 0; r < 4; ++r) {
            rb1[r] = *(const float4*)(Bp1 + (size_t)(kc + bkq + r) * ldb + n0 + bn4);
            if (DUAL)
                rb2[r] = *(const float4*)(Bp2 + (size_t)(kc + bkq + r) * ldb + n0 + bn4);
        }
    };

    auto store_regs = [&](int buf) {
        __bf16* dA  = sA  + buf * (BM * KP);
        __bf16* dB1 = sB1 + buf * (BN * KP);
        __bf16* dB2 = DUAL ? (sB2 + buf * (BN * KP)) : nullptr;
        #pragma unroll
        for (int it = 0; it < 2; ++it) {
            const int row = it * 32 + arow;
            v4bf h;
            h[0] = (__bf16)ra[it].x; h[1] = (__bf16)ra[it].y;
            h[2] = (__bf16)ra[it].z; h[3] = (__bf16)ra[it].w;
            *(v4bf*)&dA[row * KP + kswz(row, acol & 24) + (acol & 7)] = h;
        }
        // 4x4 register transpose: store 4 contiguous k values per n row (b64 stores)
        #pragma unroll
        for (int j = 0; j < 4; ++j) {
            const int row = bn4 + j;
            const int off = row * KP + kswz(row, bkq & 24) + (bkq & 7);
            v4bf h;
            h[0] = (__bf16)((const float*)&rb1[0])[j];
            h[1] = (__bf16)((const float*)&rb1[1])[j];
            h[2] = (__bf16)((const float*)&rb1[2])[j];
            h[3] = (__bf16)((const float*)&rb1[3])[j];
            *(v4bf*)&dB1[off] = h;
            if (DUAL) {
                v4bf g;
                g[0] = (__bf16)((const float*)&rb2[0])[j];
                g[1] = (__bf16)((const float*)&rb2[1])[j];
                g[2] = (__bf16)((const float*)&rb2[2])[j];
                g[3] = (__bf16)((const float*)&rb2[3])[j];
                *(v4bf*)&dB2[off] = g;
            }
        }
    };

    auto load_frag = [&](const __bf16* base, int row) -> Frag {
        Frag f;
        f.q[0] = *(const uint4*)(base + row * KP + kswz(row, kb));
        f.q[1] = *(const uint4*)(base + row * KP + kswz(row, kb + 16));
        return f;
    };

    auto compute = [&](int buf) {
        const __bf16* pA  = sA  + buf * (BM * KP);
        const __bf16* pB1 = sB1 + buf * (BN * KP);
        const __bf16* pB2 = DUAL ? (sB2 + buf * (BN * KP)) : nullptr;
        // fragments per ISA 7.12.2: lanes<16 hold K 0..7 & 16..23, lanes>=16 K 8..15 & 24..31
        Frag a[2], b1[2], b2[2];
        #pragma unroll
        for (int i = 0; i < 2; ++i) a[i] = load_frag(pA, wm0 + i * 16 + lm);
        #pragma unroll
        for (int j = 0; j < 2; ++j) b1[j] = load_frag(pB1, wn0 + j * 16 + lm);
        if (DUAL) {
            #pragma unroll
            for (int j = 0; j < 2; ++j) b2[j] = load_frag(pB2, wn0 + j * 16 + lm);
        }
        #pragma unroll
        for (int i = 0; i < 2; ++i)
            #pragma unroll
            for (int j = 0; j < 2; ++j) {
                accG[i][j] = __builtin_amdgcn_wmma_f32_16x16x32_bf16(
                    false, a[i].v, false, b1[j].v, (short)0, accG[i][j], false, false);
                if (DUAL)
                    accU[i][j] = __builtin_amdgcn_wmma_f32_16x16x32_bf16(
                        false, a[i].v, false, b2[j].v, (short)0, accU[i][j], false, false);
            }
    };

    // ---- software-pipelined main loop: global loads overlap WMMA ----
    const int nc = K / BK;
    load_regs(0);
    store_regs(0);
    __syncthreads();
    int cur = 0;
    for (int c = 0; c < nc; ++c) {
        const bool have_next = (c + 1 < nc);
        if (have_next) load_regs((c + 1) * BK);   // issue global loads first
        compute(cur);                              // WMMAs hide the load latency
        if (have_next) store_regs(cur ^ 1);        // cvt + ds_store into alternate buffer
        __syncthreads();
        cur ^= 1;
    }

    // ---- epilogue: C/D layout (lanes<16: n=lane, m=r; lanes>=16: n=lane-16, m=8+r) ----
    const int mgrp = (lane < 16) ? 0 : 8;
    #pragma unroll
    for (int i = 0; i < 2; ++i) {
        #pragma unroll
        for (int r = 0; r < 8; ++r) {
            const int mloc = wm0 + i * 16 + mgrp + r;
            const int orow = s_out[mloc];
            if (orow < 0) continue;
            const float sc = rowScale ? rowScale[s_in[mloc]] : 1.f;
            #pragma unroll
            for (int j = 0; j < 2; ++j) {
                const int ncol = n0 + wn0 + j * 16 + lm;
                const float g = accG[i][j][r];
                float val;
                if (DUAL) {
                    const float u = accU[i][j][r];
                    val = (g / (1.f + __expf(-g))) * u;       // silu(g) * u
                } else {
                    val = g * sc;
                }
                C[(size_t)orow * ldc + ncol] = val;
            }
        }
    }
}

// -------------------- final combine: out += sum_k w[t,k] * routed[slot(t,k)] --------------------
__global__ __launch_bounds__(256) void combine_kernel(
    const float* __restrict__ rt_out,
    const float* __restrict__ topk_w,
    float* __restrict__ out)
{
    const int t = blockIdx.x;
    for (int h = threadIdx.x; h < H; h += blockDim.x) {
        float s = out[(size_t)t * H + h];
        #pragma unroll
        for (int k = 0; k < TOPK; ++k)
            s += topk_w[t * TOPK + k] * rt_out[(size_t)(t * TOPK + k) * H + h];
        out[(size_t)t * H + h] = s;
    }
}

extern "C" void kernel_launch(void* const* d_in, const int* in_sizes, int n_in,
                              void* d_out, int out_size, void* d_ws, size_t ws_size,
                              hipStream_t stream)
{
    const float* x       = (const float*)d_in[0];
    const float* rw      = (const float*)d_in[1];
    const float* w_gate  = (const float*)d_in[2];
    const float* w_up    = (const float*)d_in[3];
    const float* w_down  = (const float*)d_in[4];
    const float* sh_gate = (const float*)d_in[5];
    const float* sh_up   = (const float*)d_in[6];
    const float* sh_down = (const float*)d_in[7];
    const float* sh_gw   = (const float*)d_in[8];
    float* out = (float*)d_out;

    char* ws = (char*)d_ws;
    size_t off = 0;
    auto take = [&](size_t bytes) -> char* {
        char* p = ws + off;
        off = (off + bytes + 255) & ~(size_t)255;
        return p;
    };
    float* topk_w = (float*)take((size_t)T_TOK * TOPK * sizeof(float));
    int*   topk_i = (int*)  take((size_t)T_TOK * TOPK * sizeof(int));
    float* sig    = (float*)take((size_t)T_TOK * sizeof(float));
    int*   cnt    = (int*)  take((size_t)E_EXP * sizeof(int));
    int*   l_tok  = (int*)  take((size_t)E_EXP * T_TOK * sizeof(int));
    int*   l_slot = (int*)  take((size_t)E_EXP * T_TOK * sizeof(int));
    float* act_sh = (float*)take((size_t)T_TOK * I_SH * sizeof(float));
    float* act_rt = (float*)take((size_t)T_TOK * TOPK * I_RT * sizeof(float));
    float* rt_out = (float*)take((size_t)T_TOK * TOPK * H * sizeof(float));

    hipMemsetAsync(cnt, 0, E_EXP * sizeof(int), stream);

    router_kernel<<<T_TOK / 8, 256, 0, stream>>>(x, rw, sh_gw, topk_w, topk_i, sig);
    dispatch_kernel<<<(T_TOK * TOPK) / 256, 256, 0, stream>>>(topk_i, cnt, l_tok, l_slot);

    // shared expert: act_sh = silu(x @ sh_gate) * (x @ sh_up)
    gemm_kernel<1><<<dim3(I_SH / BN, T_TOK / BM, 1), 256, 0, stream>>>(
        x, H, sh_gate, sh_up, I_SH, act_sh, I_SH, 0,
        nullptr, nullptr, 0, nullptr, T_TOK, H, nullptr);

    // routed experts: act_rt[slot] = silu(x[t] @ w_gate[e]) * (x[t] @ w_up[e])
    gemm_kernel<1><<<dim3(I_RT / BN, T_TOK / BM, E_EXP), 256, 0, stream>>>(
        x, H, w_gate, w_up, I_RT, act_rt, I_RT, (size_t)H * I_RT,
        l_tok, l_slot, T_TOK, cnt, 0, H, nullptr);

    // routed down: rt_out[slot] = act_rt[slot] @ w_down[e]
    gemm_kernel<0><<<dim3(H / BN, T_TOK / BM, E_EXP), 256, 0, stream>>>(
        act_rt, I_RT, w_down, nullptr, H, rt_out, H, (size_t)I_RT * H,
        l_slot, l_slot, T_TOK, cnt, 0, I_RT, nullptr);

    // shared down: out = (act_sh @ sh_down) * sigmoid_gate[t]   (overwrites poisoned d_out)
    gemm_kernel<0><<<dim3(H / BN, T_TOK / BM, 1), 256, 0, stream>>>(
        act_sh, I_SH, sh_down, nullptr, H, out, H, 0,
        nullptr, nullptr, 0, nullptr, T_TOK, I_SH, sig);

    combine_kernel<<<T_TOK, 256, 0, stream>>>(rt_out, topk_w, out);

    (void)in_sizes; (void)n_in; (void)out_size; (void)ws_size;
}
